// EncoderDecoderAttentionModel_2705829397052
// MI455X (gfx1250) — matrix operand order repaired and checked by reading, not compiled, not run on hardware
//
#include <hip/hip_runtime.h>

// ---------------------------------------------------------------------------
// Types for CDNA5 WMMA (wave32, v_wmma_f32_16x16x32_bf16)
// ---------------------------------------------------------------------------
typedef __bf16 bf16_t;
typedef bf16_t v16bf __attribute__((ext_vector_type(16)));
typedef bf16_t v8bf  __attribute__((ext_vector_type(8)));
typedef float  v8f   __attribute__((ext_vector_type(8)));

#define B_   64
#define T_   256
#define H_   256
#define G4H  1024
#define CVOC 32000
#define NSTEPS_ 25
#define PAD_IND_ 3

__device__ __forceinline__ float sigmf(float x) { return 1.f / (1.f + __expf(-x)); }

__device__ __forceinline__ v8f wmma_bf16(v16bf a, v16bf b, v8f c) {
  // D = A(16x32 bf16) * B(32x16 bf16) + C(16x16 f32)
  return __builtin_amdgcn_wmma_f32_16x16x32_bf16(false, a, false, b, (short)0, c,
                                                 false, false);
}

// A fragment: 16x32 (MxK) bf16, row-major A with leading dim lda.
// lane m = lane&15 ; half = lane>>4 selects K-interleave per ISA 7.12.2:
//   e0..e7  = K { 8*half + 0..7 }
//   e8..e15 = K { 16 + 8*half + 0..7 }
__device__ __forceinline__ v16bf load_a_frag(const bf16_t* __restrict__ A, int lda,
                                             int row0, int kk, int lane) {
  const int m   = row0 + (lane & 15);
  const int hlf = lane >> 4;
  const bf16_t* p = A + (size_t)m * lda + kk + 8 * hlf;
  v8bf lo = *(const v8bf*)p;
  v8bf hi = *(const v8bf*)(p + 16);
  v16bf r;
#pragma unroll
  for (int i = 0; i < 8; ++i) { r[i] = lo[i]; r[8 + i] = hi[i]; }
  return r;
}

// B fragment: 32x16 (KxN) where B = W^T and W is row-major [N, K] (torch layout).
// lane n = lane&15 ; half = lane>>4: e0..e7 = K { 8*half + 0..7 },
// e8..e15 = K { 16 + 8*half + 0..7 } (same interleave as the A fragment).
__device__ __forceinline__ v16bf load_b_fragT(const bf16_t* __restrict__ W, int ldw,
                                              int col0, int kk, int lane) {
  const int n   = col0 + (lane & 15);
  const int hlf = lane >> 4;
  const bf16_t* p = W + (size_t)n * ldw + kk + 8 * hlf;
  v8bf lo = *(const v8bf*)p;
  v8bf hi = *(const v8bf*)(p + 16);
  v16bf r;
#pragma unroll
  for (int i = 0; i < 8; ++i) { r[i] = lo[i]; r[8 + i] = hi[i]; }
  return r;
}

// Generic shared pointers on gfx12 carry the LDS byte address in the low 32
// bits; async-to-LDS VDST wants
// exactly that LDS byte address.
__device__ __forceinline__ unsigned lds_addr_u32(const void* p) {
  return (unsigned)(uintptr_t)p;
}

// ---------------------------------------------------------------------------
// Generic WMMA GEMM:  C[M,N] = A[M,K](bf16) @ W[N,K]^T(bf16) (+ bias | += C)
// block = 256 threads = 8 waves, each wave owns one 16x16 tile; all 8 waves
// share the same 16xK A tile, which is async-staged into LDS once per block
// (GLOBAL_LOAD_ASYNC_TO_LDS_B128, ASYNCcnt) and consumed via ds_load.
// grid = (N/16/8, M/16). K in {256, 512}; all dims multiples of 16/32.
// ---------------------------------------------------------------------------
__global__ void __launch_bounds__(256)
gemm_bf16_wmma(const bf16_t* __restrict__ A, int lda,
               const bf16_t* __restrict__ W, int ldw,
               const float* __restrict__ bias,
               float* __restrict__ Cf, bf16_t* __restrict__ Cb,
               long long ldc, int K, int accumulate) {
  __shared__ bf16_t sA[16 * 512];  // 16KB: block-shared A tile (max K = 512)

  const int tid  = threadIdx.x;
  const int lane = tid & 31;
  const int wave = tid >> 5;
  const int col0 = (blockIdx.x * 8 + wave) * 16;
  const int row0 = blockIdx.y * 16;
  const int n    = col0 + (lane & 15);
  const int hlf  = lane >> 4;

  // ---- async-stage the shared A tile [16 x K] into LDS (16B chunks) ----
  {
    const int chunks_per_row = K >> 3;            // 8 bf16 = 16 bytes
    const int nchunks = 16 * chunks_per_row;
    for (int c = tid; c < nchunks; c += 256) {
      const int row = c / chunks_per_row;
      const int col = (c - row * chunks_per_row) << 3;
      const bf16_t* g = A + (size_t)(row0 + row) * lda + col;
      const unsigned l = lds_addr_u32(&sA[row * K + col]);
      asm volatile("global_load_async_to_lds_b128 %0, %1, off"
                   :: "v"(l), "v"(g) : "memory");
    }
    asm volatile("s_wait_asynccnt 0x0" ::: "memory");
  }

  v8f acc;
  if (accumulate) {
#pragma unroll
    for (int r = 0; r < 8; ++r)
      acc[r] = Cf[(size_t)(row0 + r + 8 * hlf) * ldc + n];
  } else {
    const float bv = bias ? bias[n] : 0.f;
#pragma unroll
    for (int r = 0; r < 8; ++r) acc[r] = bv;
  }
  __syncthreads();  // A tile visible to all waves

  for (int kk = 0; kk < K; kk += 32) {
    if (kk + 32 < K)
      __builtin_prefetch((const void*)(W + (size_t)n * ldw + kk + 32), 0, 3);
    v16bf a = load_a_frag(sA, K, 0, kk, lane);     // LDS (ds_load_b128)
    v16bf b = load_b_fragT(W, ldw, col0, kk, lane);// global, L2-resident weights
    acc = wmma_bf16(a, b, acc);
  }

#pragma unroll
  for (int r = 0; r < 8; ++r) {
    const size_t idx = (size_t)(row0 + r + 8 * hlf) * ldc + n;
    if (Cf) Cf[idx] = acc[r];
    if (Cb) Cb[idx] = (bf16_t)acc[r];
  }
}

// ---------------------------------------------------------------------------
// Persistent single-WGP encoder LSTM layer. 1024 threads = 32 waves.
// LDS: h[64][256] bf16 (32KB) + c[64][256] f32 (64KB) + gates[64][1024] bf16
// (128KB) = 224KB of the 320KB WGP LDS.
// Per step: gates = Gx[t] + h_{t-1} @ Whh^T  (WMMA, A-frags from LDS),
// then cell update with per-batch valid-mask freezing (pack_padded semantics).
// ---------------------------------------------------------------------------
__global__ void __launch_bounds__(1024)
enc_layer_kernel(const bf16_t* __restrict__ Gx,   // [T,B,4H] = x@Wih^T + (bih+bhh)
                 const bf16_t* __restrict__ Whh,  // [4H,H] row-major
                 const int* __restrict__ lengths, // [B]
                 bf16_t* __restrict__ Hout,       // [T,B,H] zero-padded
                 bf16_t* __restrict__ h_final,    // [B,H] bf16 (decoder init)
                 float* __restrict__ c_final) {   // [B,H] f32  (decoder init)
  extern __shared__ char smem[];
  bf16_t* sh_h = (bf16_t*)smem;                               // [64][256]
  float*  sh_c = (float*)(smem + B_ * H_ * 2);                // [64][256]
  bf16_t* sh_g = (bf16_t*)(smem + B_ * H_ * 2 + B_ * H_ * 4); // [64][1024]

  const int tid  = threadIdx.x;
  const int lane = tid & 31;
  const int wave = tid >> 5;   // 0..31
  const int mt   = wave >> 3;  // A row tile (0..3)
  const int ntg  = wave & 7;   // N tile group (0..7)
  const int hlf  = lane >> 4;
  const int ncol = lane & 15;

  for (int i = tid; i < B_ * H_; i += 1024) { sh_h[i] = (bf16_t)0.f; sh_c[i] = 0.f; }
  __syncthreads();

  for (int t = 0; t < T_; ++t) {
    const bf16_t* gx_t = Gx + (size_t)t * B_ * G4H;
    // ---- h @ Whh^T : 8 tiles per wave (4 Mtiles x 64 Ntiles total) ----
    for (int j = 0; j < 8; ++j) {
      const int col0 = (ntg * 8 + j) * 16;
      const int n    = col0 + ncol;
      v8f acc;
#pragma unroll
      for (int r = 0; r < 8; ++r)
        acc[r] = (float)gx_t[(size_t)(mt * 16 + r + 8 * hlf) * G4H + n];
#pragma unroll
      for (int k8 = 0; k8 < 8; ++k8) {
        v16bf a = load_a_frag(sh_h, H_, mt * 16, k8 * 32, lane);
        v16bf b = load_b_fragT(Whh, H_, col0, k8 * 32, lane);
        acc = wmma_bf16(a, b, acc);
      }
#pragma unroll
      for (int r = 0; r < 8; ++r)
        sh_g[(size_t)(mt * 16 + r + 8 * hlf) * G4H + n] = (bf16_t)acc[r];
    }
    __syncthreads();

    // ---- cell update + freeze/zero-pad ----
    for (int i = tid; i < B_ * H_; i += 1024) {
      const int b = i >> 8, j = i & 255;
      const float gi = (float)sh_g[(size_t)b * G4H + j];
      const float gf = (float)sh_g[(size_t)b * G4H + 256 + j];
      const float gg = (float)sh_g[(size_t)b * G4H + 512 + j];
      const float go = (float)sh_g[(size_t)b * G4H + 768 + j];
      const float c_old = sh_c[i];
      const float c2 = sigmf(gf) * c_old + sigmf(gi) * tanhf(gg);
      const float h2 = sigmf(go) * tanhf(c2);
      const bool valid = (t < lengths[b]);
      const float h_old = (float)sh_h[i];
      sh_c[i] = valid ? c2 : c_old;
      sh_h[i] = (bf16_t)(valid ? h2 : h_old);
      Hout[(size_t)t * B_ * H_ + i] = (bf16_t)(valid ? h2 : 0.f);
    }
    __syncthreads();
  }

  for (int i = tid; i < B_ * H_; i += 1024) {
    h_final[i] = sh_h[i];
    c_final[i] = sh_c[i];
  }
}

// ---------------------------------------------------------------------------
// Small helper kernels
// ---------------------------------------------------------------------------
__global__ void cvt_f32_bf16(const float* __restrict__ s, bf16_t* __restrict__ d, int n) {
  const int i = blockIdx.x * 256 + threadIdx.x;
  if (i < n) d[i] = (bf16_t)s[i];
}

__global__ void bias_sum_kernel(const float* __restrict__ a, const float* __restrict__ b,
                                float* __restrict__ o, int n) {
  const int i = blockIdx.x * 256 + threadIdx.x;
  if (i < n) o[i] = a[i] + b[i];
}

__global__ void __launch_bounds__(256)
lengths_kernel(const int* __restrict__ x, int* __restrict__ lengths) {
  const int b = blockIdx.x, tid = threadIdx.x;
  __shared__ int red[256];
  red[tid] = (x[b * T_ + tid] != PAD_IND_) ? 1 : 0;
  __syncthreads();
  for (int o = 128; o > 0; o >>= 1) {
    if (tid < o) red[tid] += red[tid + o];
    __syncthreads();
  }
  if (tid == 0) lengths[b] = red[0];
}

// Xemb[t][b][:] = bf16(enc_embed[x[b][t]][:]) ; grid = T*B blocks of 256
__global__ void gather_enc_kernel(const int* __restrict__ x, const float* __restrict__ emb,
                                  bf16_t* __restrict__ Xemb) {
  const int blk = blockIdx.x;
  const int t = blk >> 6, b = blk & 63, d = threadIdx.x;
  const int tok = x[b * T_ + t];
  Xemb[(size_t)blk * H_ + d] = (bf16_t)emb[(size_t)tok * H_ + d];
}

__global__ void init_tok_kernel(int* __restrict__ tok) {
  if (threadIdx.x < B_) tok[threadIdx.x] = 0;  // SOS_IND
}

// hsum = h0 + h1 + h2 (decoder state sum feeding the query projection)
__global__ void hsum_kernel(const bf16_t* __restrict__ h, bf16_t* __restrict__ hs) {
  const int i = blockIdx.x * 256 + threadIdx.x;
  hs[i] = (bf16_t)((float)h[i] + (float)h[B_ * H_ + i] + (float)h[2 * B_ * H_ + i]);
}

// scores -> softmax (full T, matching reference: no length masking) -> context
__global__ void __launch_bounds__(256)
attn_ctx_kernel(const float* __restrict__ q, const bf16_t* __restrict__ kp,
                const bf16_t* __restrict__ vp, bf16_t* __restrict__ ctx_raw) {
  const int b = blockIdx.x;
  const int tid = threadIdx.x;  // role 1: t index ; role 2: h index
  __shared__ float attn[256];
  __shared__ float red[256];

  const float*  qr = q  + (size_t)b * H_;
  const bf16_t* kr = kp + (size_t)tid * B_ * H_ + (size_t)b * H_;
  float s = 0.f;
  for (int h = 0; h < H_; ++h) s += qr[h] * (float)kr[h];
  s *= 0.0625f;  // 1/sqrt(256)

  red[tid] = s; __syncthreads();
  for (int o = 128; o > 0; o >>= 1) { if (tid < o) red[tid] = fmaxf(red[tid], red[tid + o]); __syncthreads(); }
  const float mx = red[0];
  __syncthreads();
  const float e = __expf(s - mx);
  red[tid] = e; __syncthreads();
  for (int o = 128; o > 0; o >>= 1) { if (tid < o) red[tid] += red[tid + o]; __syncthreads(); }
  const float inv = 1.f / red[0];
  __syncthreads();
  attn[tid] = e * inv;
  __syncthreads();

  float acc = 0.f;
  for (int t = 0; t < T_; ++t)
    acc += attn[t] * (float)vp[(size_t)t * B_ * H_ + (size_t)b * H_ + tid];
  ctx_raw[(size_t)b * H_ + tid] = (bf16_t)acc;
}

// z = [ dec_embed[tok] ; ctx ]  (bf16, [B, 2H])
__global__ void build_z_kernel(const int* __restrict__ tok, const float* __restrict__ dec_embed,
                               const bf16_t* __restrict__ ctx, bf16_t* __restrict__ z) {
  const int b = blockIdx.x, j = threadIdx.x;
  z[(size_t)b * 512 + j]       = (bf16_t)dec_embed[(size_t)tok[b] * H_ + j];
  z[(size_t)b * 512 + 256 + j] = ctx[(size_t)b * H_ + j];
}

// LSTM cell pointwise for the decoder (gates precomputed in f32)
__global__ void dec_cell_kernel(const float* __restrict__ gates,
                                bf16_t* __restrict__ h, float* __restrict__ c) {
  const int i = blockIdx.x * 256 + threadIdx.x;  // 64*256
  const int b = i >> 8, j = i & 255;
  const float gi = gates[(size_t)b * G4H + j];
  const float gf = gates[(size_t)b * G4H + 256 + j];
  const float gg = gates[(size_t)b * G4H + 512 + j];
  const float go = gates[(size_t)b * G4H + 768 + j];
  const float c2 = sigmf(gf) * c[i] + sigmf(gi) * tanhf(gg);
  h[i] = (bf16_t)(sigmf(go) * tanhf(c2));
  c[i] = c2;
}

// Greedy argmax over C=32000, lowest index wins ties (jnp.argmax semantics)
__global__ void __launch_bounds__(256)
argmax_kernel(const float* __restrict__ logits, long long ldb, int* __restrict__ tok) {
  const int b = blockIdx.x, tid = threadIdx.x;
  const float* row = logits + (size_t)b * ldb;
  float best = -3.4e38f; int bi = 0x7fffffff;
  for (int c = tid; c < CVOC; c += 256) {
    const float v = row[c];
    if (v > best || (v == best && c < bi)) { best = v; bi = c; }
  }
  __shared__ float bv[256]; __shared__ int bx[256];
  bv[tid] = best; bx[tid] = bi; __syncthreads();
  for (int o = 128; o > 0; o >>= 1) {
    if (tid < o) {
      if (bv[tid + o] > bv[tid] || (bv[tid + o] == bv[tid] && bx[tid + o] < bx[tid])) {
        bv[tid] = bv[tid + o]; bx[tid] = bx[tid + o];
      }
    }
    __syncthreads();
  }
  if (tid == 0) tok[b] = bx[0];
}

// ---------------------------------------------------------------------------
// Host orchestration
// ---------------------------------------------------------------------------
extern "C" void kernel_launch(void* const* d_in, const int* in_sizes, int n_in,
                              void* d_out, int out_size, void* d_ws, size_t ws_size,
                              hipStream_t stream) {
  (void)in_sizes; (void)n_in; (void)out_size; (void)ws_size;

  const int*   x         = (const int*)  d_in[0];
  const float* enc_embed = (const float*)d_in[2];
  const float* enc_Wih   = (const float*)d_in[3];
  const float* enc_Whh   = (const float*)d_in[4];
  const float* enc_bih   = (const float*)d_in[5];
  const float* enc_bhh   = (const float*)d_in[6];
  const float* Wq = (const float*)d_in[7];  const float* bq = (const float*)d_in[8];
  const float* Wk = (const float*)d_in[9];  const float* bk = (const float*)d_in[10];
  const float* Wv = (const float*)d_in[11]; const float* bv = (const float*)d_in[12];
  const float* Wo = (const float*)d_in[13]; const float* bo = (const float*)d_in[14];
  const float* dec_embed  = (const float*)d_in[15];
  const float* dec_Wih0   = (const float*)d_in[16];
  const float* dec_WihR   = (const float*)d_in[17];
  const float* dec_Whh    = (const float*)d_in[18];
  const float* dec_bih    = (const float*)d_in[19];
  const float* dec_bhh    = (const float*)d_in[20];
  const float* dec_W      = (const float*)d_in[21];
  const float* dec_b      = (const float*)d_in[22];

  // ---- workspace carve-out ----
  char* base = (char*)d_ws;
  size_t off = 0;
  auto alloc = [&](size_t bytes) -> void* {
    void* p = base + off;
    off = (off + bytes + 255) & ~(size_t)255;
    return p;
  };
  bf16_t* eWih_b  = (bf16_t*)alloc(3 * G4H * H_ * sizeof(bf16_t));
  bf16_t* eWhh_b  = (bf16_t*)alloc(3 * G4H * H_ * sizeof(bf16_t));
  bf16_t* Wq_b    = (bf16_t*)alloc(H_ * H_ * sizeof(bf16_t));
  bf16_t* Wk_b    = (bf16_t*)alloc(H_ * H_ * sizeof(bf16_t));
  bf16_t* Wv_b    = (bf16_t*)alloc(H_ * H_ * sizeof(bf16_t));
  bf16_t* Wo_b    = (bf16_t*)alloc(H_ * H_ * sizeof(bf16_t));
  bf16_t* dWih0_b = (bf16_t*)alloc(G4H * 512 * sizeof(bf16_t));
  bf16_t* dWihR_b = (bf16_t*)alloc(2 * G4H * H_ * sizeof(bf16_t));
  bf16_t* dWhh_b  = (bf16_t*)alloc(3 * G4H * H_ * sizeof(bf16_t));
  bf16_t* dW_b    = (bf16_t*)alloc((size_t)CVOC * H_ * sizeof(bf16_t));
  float*  ebsum   = (float*) alloc(3 * G4H * sizeof(float));
  float*  dbsum   = (float*) alloc(3 * G4H * sizeof(float));
  int*    lengths = (int*)   alloc(B_ * sizeof(int));
  bf16_t* Xemb    = (bf16_t*)alloc((size_t)T_ * B_ * H_ * sizeof(bf16_t));
  bf16_t* Gx      = (bf16_t*)alloc((size_t)T_ * B_ * G4H * sizeof(bf16_t));
  bf16_t* HoutA   = (bf16_t*)alloc((size_t)T_ * B_ * H_ * sizeof(bf16_t));
  bf16_t* HoutB   = (bf16_t*)alloc((size_t)T_ * B_ * H_ * sizeof(bf16_t));
  bf16_t* kp      = (bf16_t*)alloc((size_t)T_ * B_ * H_ * sizeof(bf16_t));
  bf16_t* vp      = (bf16_t*)alloc((size_t)T_ * B_ * H_ * sizeof(bf16_t));
  bf16_t* dec_h   = (bf16_t*)alloc(3 * B_ * H_ * sizeof(bf16_t));
  float*  dec_c   = (float*) alloc(3 * B_ * H_ * sizeof(float));
  bf16_t* hsumb   = (bf16_t*)alloc(B_ * H_ * sizeof(bf16_t));
  float*  qf      = (float*) alloc(B_ * H_ * sizeof(float));
  bf16_t* ctxraw  = (bf16_t*)alloc(B_ * H_ * sizeof(bf16_t));
  bf16_t* ctxb    = (bf16_t*)alloc(B_ * H_ * sizeof(bf16_t));
  bf16_t* zb      = (bf16_t*)alloc(B_ * 512 * sizeof(bf16_t));
  float*  gates   = (float*) alloc(B_ * G4H * sizeof(float));
  int*    tok     = (int*)   alloc(B_ * sizeof(int));

  auto cvt = [&](const float* s, bf16_t* d, int n) {
    cvt_f32_bf16<<<(n + 255) / 256, 256, 0, stream>>>(s, d, n);
  };

  // ---- one-time prep ----
  cvt(enc_Wih, eWih_b, 3 * G4H * H_);
  cvt(enc_Whh, eWhh_b, 3 * G4H * H_);
  cvt(Wq, Wq_b, H_ * H_); cvt(Wk, Wk_b, H_ * H_);
  cvt(Wv, Wv_b, H_ * H_); cvt(Wo, Wo_b, H_ * H_);
  cvt(dec_Wih0, dWih0_b, G4H * 512);
  cvt(dec_WihR, dWihR_b, 2 * G4H * H_);
  cvt(dec_Whh, dWhh_b, 3 * G4H * H_);
  cvt(dec_W, dW_b, CVOC * H_);
  bias_sum_kernel<<<12, 256, 0, stream>>>(enc_bih, enc_bhh, ebsum, 3 * G4H);
  bias_sum_kernel<<<12, 256, 0, stream>>>(dec_bih, dec_bhh, dbsum, 3 * G4H);
  lengths_kernel<<<B_, 256, 0, stream>>>(x, lengths);
  gather_enc_kernel<<<T_ * B_, 256, 0, stream>>>(x, enc_embed, Xemb);

  // ---- encoder: 3 layers, input GEMM hoisted, recurrence on one WGP ----
  const size_t encLDS = (size_t)B_ * H_ * 2 + (size_t)B_ * H_ * 4 + (size_t)B_ * G4H * 2; // 224KB
  const bf16_t* Xin = Xemb;
  bf16_t* outs[3] = {HoutA, HoutB, HoutA};
  for (int l = 0; l < 3; ++l) {
    // Gx[T*B, 4H] = Xin @ Wih_l^T + (bih+bhh)_l
    gemm_bf16_wmma<<<dim3(G4H / 16 / 8, (T_ * B_) / 16), 256, 0, stream>>>(
        Xin, H_, eWih_b + (size_t)l * G4H * H_, H_, ebsum + l * G4H,
        nullptr, Gx, G4H, H_, 0);
    enc_layer_kernel<<<1, 1024, encLDS, stream>>>(
        Gx, eWhh_b + (size_t)l * G4H * H_, lengths, outs[l],
        dec_h + (size_t)l * B_ * H_, dec_c + (size_t)l * B_ * H_);
    Xin = outs[l];
  }
  const bf16_t* enc_hid = HoutA;  // [T,B,H], zero-padded past lengths

  // ---- attention K/V projections (loop-invariant) ----
  gemm_bf16_wmma<<<dim3(H_ / 16 / 8, (T_ * B_) / 16), 256, 0, stream>>>(
      enc_hid, H_, Wk_b, H_, bk, nullptr, kp, H_, H_, 0);
  gemm_bf16_wmma<<<dim3(H_ / 16 / 8, (T_ * B_) / 16), 256, 0, stream>>>(
      enc_hid, H_, Wv_b, H_, bv, nullptr, vp, H_, H_, 0);

  // ---- greedy decode: 25 steps ----
  init_tok_kernel<<<1, 64, 0, stream>>>(tok);
  const long long outLd = (long long)NSTEPS_ * CVOC;  // row stride for batch b in d_out
  for (int step = 0; step < NSTEPS_; ++step) {
    hsum_kernel<<<B_, 256, 0, stream>>>(dec_h, hsumb);
    gemm_bf16_wmma<<<dim3(H_ / 16 / 8, B_ / 16), 256, 0, stream>>>(
        hsumb, H_, Wq_b, H_, bq, qf, nullptr, H_, H_, 0);
    attn_ctx_kernel<<<B_, 256, 0, stream>>>(qf, kp, vp, ctxraw);
    gemm_bf16_wmma<<<dim3(H_ / 16 / 8, B_ / 16), 256, 0, stream>>>(
        ctxraw, H_, Wo_b, H_, bo, nullptr, ctxb, H_, H_, 0);
    build_z_kernel<<<B_, 256, 0, stream>>>(tok, dec_embed, ctxb, zb);

    for (int l = 0; l < 3; ++l) {
      const bf16_t* zin = (l == 0) ? zb : dec_h + (size_t)(l - 1) * B_ * H_;
      const int Kin = (l == 0) ? 512 : H_;
      const bf16_t* Wih = (l == 0) ? dWih0_b : dWihR_b + (size_t)(l - 1) * G4H * H_;
      // gates = z @ Wih^T + (bih+bhh)
      gemm_bf16_wmma<<<dim3(G4H / 16 / 8, B_ / 16), 256, 0, stream>>>(
          zin, Kin, Wih, Kin, dbsum + l * G4H, gates, nullptr, G4H, Kin, 0);
      // gates += h_l @ Whh^T
      gemm_bf16_wmma<<<dim3(G4H / 16 / 8, B_ / 16), 256, 0, stream>>>(
          dec_h + (size_t)l * B_ * H_, H_, dWhh_b + (size_t)l * G4H * H_, H_,
          nullptr, gates, nullptr, G4H, H_, 1);
      dec_cell_kernel<<<B_, 256, 0, stream>>>(gates, dec_h + (size_t)l * B_ * H_,
                                              dec_c + (size_t)l * B_ * H_);
    }

    // logits straight into d_out[b][step][0][:]  (row stride 25*C)
    float* logits = (float*)d_out + (size_t)step * CVOC;
    gemm_bf16_wmma<<<dim3(CVOC / 16 / 8, B_ / 16), 256, 0, stream>>>(
        dec_h + (size_t)2 * B_ * H_, H_, dW_b, H_, dec_b, logits, nullptr,
        outLd, H_, 0);
    argmax_kernel<<<B_, 256, 0, stream>>>(logits, outLd, tok);
  }
}
